// GATv2_76725295775762
// MI455X (gfx1250) — matrix operand order repaired and checked by reading, not compiled, not run on hardware
//
#include <hip/hip_runtime.h>
#include <hip/hip_bf16.h>

// ---------------------------------------------------------------------------
// GATv2 3-layer pipeline for MI455X (gfx1250, wave32, WMMA).
// N=50000, E=400000 (+N self loops), DIN=128, H=4, C=64, HC=256, OUT=40.
// ---------------------------------------------------------------------------

#define Nn   50000
#define Ee   400000
#define ET   450000      // E + N self loops
#define DIN  128
#define Hh   4
#define Cc0  64
#define HC   256
#define OUTW 40
#define NEG_SLOPE 0.2f
#define BN_EPS    1e-5f

typedef __attribute__((ext_vector_type(16))) __bf16 v16bf;
typedef __attribute__((ext_vector_type(8)))  __bf16 v8bf;
typedef __attribute__((ext_vector_type(8)))  float  v8f;

// fp32 -> bf16 round-to-nearest-even (only used in one-time conversion passes)
__device__ __forceinline__ __bf16 f2bf(float f) {
    unsigned u = __float_as_uint(f);
    unsigned r = (u + 0x7FFFu + ((u >> 16) & 1u)) >> 16;
    unsigned short s = (unsigned short)r;
    __bf16 o;
    __builtin_memcpy(&o, &s, 2);
    return o;
}

__device__ __forceinline__ void atomicMaxFloat(float* addr, float v) {
    if (v >= 0.0f) atomicMax((int*)addr, __float_as_int(v));
    else           atomicMin((unsigned int*)addr, __float_as_uint(v));
}

// ---------------------------------------------------------------------------
// One-time conversion passes (keep the GEMM inner loop conversion-free)
// ---------------------------------------------------------------------------
__global__ void cvt_a_bf16(const float* __restrict__ x, __bf16* __restrict__ y,
                           size_t n) {
    size_t i = (size_t)blockIdx.x * blockDim.x + threadIdx.x;
    if (i < n) y[i] = f2bf(x[i]);
}

// Wt[m*K + k] = bf16(W[k*M + m])   (column-major weight, contiguous per column)
__global__ void cvt_w_bf16_t(const float* __restrict__ W, __bf16* __restrict__ Wt,
                             int K, int M) {
    int idx = blockIdx.x * blockDim.x + threadIdx.x;
    if (idx >= K * M) return;
    int k = idx / M, m = idx % M;            // coalesced read of W
    Wt[(size_t)m * K + k] = f2bf(W[idx]);
}

// ---------------------------------------------------------------------------
// WMMA GEMM:  C[N x M] = A[N x K] @ W[K x M] + b1 [+ b2]
//   A  : bf16 row-major  (pre-converted)
//   Wt : bf16 [M x K]    (pre-converted, transposed)
// One wave32 per 16x64 strip (4 col-tiles, A fragment reused 4x).
// All fragment loads are contiguous 16B (global_load_b128).
// Partial tiles handled by address clamping (no EXEC predication):
// garbage in B column n only affects D column n, which is never stored.
// ---------------------------------------------------------------------------
__global__ void wmma_gemm_bias(const __bf16* __restrict__ A,
                               const __bf16* __restrict__ Wt,
                               const float* __restrict__ b1,
                               const float* __restrict__ b2,
                               float* __restrict__ C,
                               int K, int M, int G /* col groups of 64 */) {
    const int wave  = blockIdx.x * (blockDim.x >> 5) + (threadIdx.x >> 5);
    const int total = (Nn / 16) * G;
    if (wave >= total) return;                 // wave-uniform: EXEC stays all-ones

    const int tm   = wave / G;
    const int tg   = wave % G;
    const int lane = threadIdx.x & 31;
    const int half = lane >> 4;                // 0: lanes 0-15, 1: lanes 16-31
    const int lan  = lane & 15;

    const int arow = tm * 16 + lan;

    v8f acc0 = {0.f,0.f,0.f,0.f,0.f,0.f,0.f,0.f};
    v8f acc1 = acc0, acc2 = acc0, acc3 = acc0;

    // clamped column addresses for the 4 tiles (loads always in-bounds)
    int colv[4], colc[4];
    #pragma unroll
    for (int t = 0; t < 4; ++t) {
        colv[t] = (tg * 4 + t) * 16 + lan;
        colc[t] = colv[t] < M ? colv[t] : (M - 1);
    }

    const __bf16* Ab = A + (size_t)arow * K + 8 * half;   // lane's A base
    for (int k0 = 0; k0 < K; k0 += 32) {
        // A fragment: elements 0..7 = K{8*half+0..7}, 8..15 = K{16+8*half+0..7}
        v8bf alo = *(const v8bf*)(Ab + k0);
        v8bf ahi = *(const v8bf*)(Ab + k0 + 16);
        v16bf af = __builtin_shufflevector(alo, ahi,
                     0,1,2,3,4,5,6,7,8,9,10,11,12,13,14,15);

        #pragma unroll
        for (int t = 0; t < 4; ++t) {
            // B fragment: lane's column, elements 0..7 = K{8*half+0..7}, 8..15 = K{16+8*half+0..7}
            const __bf16* Bb = Wt + (size_t)colc[t] * K + k0 + 8 * half;
            v8bf blo = *(const v8bf*)(Bb);
            v8bf bhi = *(const v8bf*)(Bb + 16);
            v16bf bf = __builtin_shufflevector(blo, bhi,
                         0,1,2,3,4,5,6,7,8,9,10,11,12,13,14,15);
            v8f* accp = (t == 0) ? &acc0 : (t == 1) ? &acc1 : (t == 2) ? &acc2 : &acc3;
            *accp = __builtin_amdgcn_wmma_f32_16x16x32_bf16(
                        false, af, false, bf, (short)0, *accp, false, false);
        }
    }

    #pragma unroll
    for (int t = 0; t < 4; ++t) {
        int col = colv[t];
        if (col < M) {
            v8f acc = (t == 0) ? acc0 : (t == 1) ? acc1 : (t == 2) ? acc2 : acc3;
            float bias = b1[col] + (b2 ? b2[col] : 0.f);
            #pragma unroll
            for (int r = 0; r < 8; ++r) {
                int orow = tm * 16 + r + half * 8;
                C[(size_t)orow * M + col] = acc[r] + bias;
            }
        }
    }
}

// ---------------------------------------------------------------------------
__global__ void init_mz(float* __restrict__ mb, float* __restrict__ zb, int n) {
    int i = blockIdx.x * blockDim.x + threadIdx.x;
    if (i < n) { mb[i] = -__builtin_inff(); zb[i] = 0.f; }
}

// Per (edge, head): logit = sum_c leakyrelu(xl[s]+xr[d])*att ; atomicMax(m[d])
__global__ void edge_logit(const float* __restrict__ xl,
                           const float* __restrict__ xr,
                           const float* __restrict__ att,
                           const int* __restrict__ src,
                           const int* __restrict__ dst,
                           float* __restrict__ logit,
                           float* __restrict__ mb, int C) {
    const int wid = blockIdx.x * (blockDim.x >> 5) + (threadIdx.x >> 5);
    if (wid >= ET * Hh) return;
    const int lane = threadIdx.x & 31;
    const int e = wid / Hh, h = wid % Hh;
    const int s = (e < Ee) ? src[e] : (e - Ee);
    const int d = (e < Ee) ? dst[e] : (e - Ee);

    const float* lp = xl + (size_t)s * (Hh * C) + h * C;
    const float* rp = xr + (size_t)d * (Hh * C) + h * C;
    const float* ap = att + h * C;

    float acc = 0.f;
    for (int c = lane; c < C; c += 32) {
        float v = lp[c] + rp[c];
        v = (v > 0.f) ? v : v * NEG_SLOPE;
        acc += v * ap[c];
    }
    #pragma unroll
    for (int off = 16; off > 0; off >>= 1) acc += __shfl_down(acc, off, 32);
    if (lane == 0) {
        logit[wid] = acc;
        atomicMaxFloat(&mb[(size_t)d * Hh + h], acc);
    }
}

// a = exp(logit - m[d]); z[d] += a
__global__ void edge_exp(const float* __restrict__ logit,
                         const float* __restrict__ mb,
                         float* __restrict__ zb,
                         float* __restrict__ expv,
                         const int* __restrict__ dst) {
    int id = blockIdx.x * blockDim.x + threadIdx.x;
    if (id >= ET * Hh) return;
    int e = id / Hh, h = id % Hh;
    int d = (e < Ee) ? dst[e] : (e - Ee);
    float a = expf(logit[id] - mb[(size_t)d * Hh + h]);
    expv[id] = a;
    atomicAdd(&zb[(size_t)d * Hh + h], a);
}

// agg[d] += (a/z) * scale * xl[s]   (fold=1: head-mean into width-C output)
__global__ void edge_scatter(const float* __restrict__ xl,
                             const float* __restrict__ expv,
                             const float* __restrict__ zb,
                             const int* __restrict__ src,
                             const int* __restrict__ dst,
                             float* __restrict__ agg,
                             int C, int outW, int fold, float scale) {
    const int wid = blockIdx.x * (blockDim.x >> 5) + (threadIdx.x >> 5);
    if (wid >= ET * Hh) return;
    const int lane = threadIdx.x & 31;
    const int e = wid / Hh, h = wid % Hh;
    const int s = (e < Ee) ? src[e] : (e - Ee);
    const int d = (e < Ee) ? dst[e] : (e - Ee);

    const float alpha = expv[wid] / zb[(size_t)d * Hh + h] * scale;
    const float* xrow = xl + (size_t)s * (Hh * C) + h * C;
    float* arow = agg + (size_t)d * outW + (fold ? 0 : h * C);
    for (int c = lane; c < C; c += 32)
        atomicAdd(&arow[c], alpha * xrow[c]);
}

// Column mean / biased variance over N rows (one block per column)
__global__ void col_stats(const float* __restrict__ X, int M,
                          float* __restrict__ mu, float* __restrict__ var) {
    const int col = blockIdx.x;
    float s = 0.f, s2 = 0.f;
    for (int r = threadIdx.x; r < Nn; r += blockDim.x) {
        float v = X[(size_t)r * M + col];
        s += v; s2 += v * v;
    }
    __shared__ float sh[256], sh2[256];
    sh[threadIdx.x] = s; sh2[threadIdx.x] = s2;
    __syncthreads();
    for (int o = blockDim.x >> 1; o > 0; o >>= 1) {
        if (threadIdx.x < o) {
            sh[threadIdx.x]  += sh[threadIdx.x + o];
            sh2[threadIdx.x] += sh2[threadIdx.x + o];
        }
        __syncthreads();
    }
    if (threadIdx.x == 0) {
        float m = sh[0] / (float)Nn;
        mu[col]  = m;
        var[col] = sh2[0] / (float)Nn - m * m;   // biased, matches x.var(0)
    }
}

__global__ void bn_relu(const float* __restrict__ x,
                        const float* __restrict__ g,
                        const float* __restrict__ b,
                        const float* __restrict__ mu,
                        const float* __restrict__ var,
                        float* __restrict__ y, int M) {
    size_t i = (size_t)blockIdx.x * blockDim.x + threadIdx.x;
    if (i >= (size_t)Nn * M) return;
    int col = (int)(i % M);
    float v = g[col] * (x[i] - mu[col]) * rsqrtf(var[col] + BN_EPS) + b[col];
    y[i] = v > 0.f ? v : 0.f;
}

// ---------------------------------------------------------------------------
extern "C" void kernel_launch(void* const* d_in, const int* in_sizes, int n_in,
                              void* d_out, int out_size, void* d_ws, size_t ws_size,
                              hipStream_t stream) {
    (void)in_sizes; (void)n_in; (void)out_size; (void)ws_size;

    const float* x    = (const float*)d_in[0];
    const int*   src  = (const int*)  d_in[1];
    const int*   dst  = (const int*)  d_in[2];
    const float* Wl0  = (const float*)d_in[3];  const float* bl0 = (const float*)d_in[4];
    const float* Wr0  = (const float*)d_in[5];  const float* br0 = (const float*)d_in[6];
    const float* att0 = (const float*)d_in[7];  const float* bc0 = (const float*)d_in[8];
    const float* Ws0  = (const float*)d_in[9];  const float* bs0 = (const float*)d_in[10];
    const float* g0   = (const float*)d_in[11]; const float* be0 = (const float*)d_in[12];
    const float* Wl1  = (const float*)d_in[13]; const float* bl1 = (const float*)d_in[14];
    const float* Wr1  = (const float*)d_in[15]; const float* br1 = (const float*)d_in[16];
    const float* att1 = (const float*)d_in[17]; const float* bc1 = (const float*)d_in[18];
    const float* Ws1  = (const float*)d_in[19]; const float* bs1 = (const float*)d_in[20];
    const float* g1   = (const float*)d_in[21]; const float* be1 = (const float*)d_in[22];
    const float* Wl2  = (const float*)d_in[23]; const float* bl2 = (const float*)d_in[24];
    const float* Wr2  = (const float*)d_in[25]; const float* br2 = (const float*)d_in[26];
    const float* att2 = (const float*)d_in[27]; const float* bc2 = (const float*)d_in[28];
    const float* Ws2  = (const float*)d_in[29]; const float* bs2 = (const float*)d_in[30];

    // ---- workspace layout ----
    float* ws  = (float*)d_ws;
    float* XL  = ws;                         // N*HC  (layer2: N*160)
    float* XR  = XL  + (size_t)Nn * HC;      // N*HC
    float* X1  = XR  + (size_t)Nn * HC;      // N*HC
    float* X2  = X1  + (size_t)Nn * HC;      // N*HC (also layer0 agg buffer)
    float* LOG = X2  + (size_t)Nn * HC;      // ET*H
    float* EXPV= LOG + (size_t)ET * Hh;      // ET*H
    float* MB  = EXPV+ (size_t)ET * Hh;      // N*H
    float* ZB  = MB  + (size_t)Nn * Hh;      // N*H
    float* MU  = ZB  + (size_t)Nn * Hh;      // HC
    float* VAR = MU  + HC;                   // HC
    __bf16* ABF = (__bf16*)(VAR + HC);       // N*HC bf16 activations
    __bf16* WT  = ABF + (size_t)Nn * HC;     // 256*256 bf16 transposed weight

    float* Hout = (float*)d_out;             // [N, 256]  pre-BN layer-1 result
    float* Oout = Hout + (size_t)Nn * HC;    // [N, 40]   final output

    auto cvtA = [&](const float* src_f, int K) {
        size_t n = (size_t)Nn * K;
        cvt_a_bf16<<<(int)((n + 255) / 256), 256, 0, stream>>>(src_f, ABF, n);
    };
    // convert weight + run GEMM (stream order serializes WT reuse)
    auto gemm = [&](const float* W, const float* b1, const float* b2,
                    float* C, int K, int M) {
        int kw = K * M;
        cvt_w_bf16_t<<<(kw + 255) / 256, 256, 0, stream>>>(W, WT, K, M);
        int G = ((M + 15) / 16 + 3) / 4;     // 64-wide col groups per wave
        int tiles = (Nn / 16) * G;
        int blocks = (tiles + 7) / 8;        // 8 waves per 256-thread block
        wmma_gemm_bias<<<blocks, 256, 0, stream>>>(ABF, WT, b1, b2, C, K, M, G);
    };

    auto attention = [&](const float* xl, const float* xr, const float* att,
                         float* agg, int C, int outW, int fold, float scale) {
        int nm = Nn * Hh;
        init_mz<<<(nm + 255) / 256, 256, 0, stream>>>(MB, ZB, nm);
        int wblocks = (ET * Hh + 7) / 8;     // one wave per (edge, head)
        edge_logit<<<wblocks, 256, 0, stream>>>(xl, xr, att, src, dst, LOG, MB, C);
        edge_exp<<<(ET * Hh + 255) / 256, 256, 0, stream>>>(LOG, MB, ZB, EXPV, dst);
        edge_scatter<<<wblocks, 256, 0, stream>>>(xl, EXPV, ZB, src, dst, agg,
                                                  C, outW, fold, scale);
    };

    const size_t elemsHC = (size_t)Nn * HC;

    // ================= Layer 0:  x[128] -> x1[256] =================
    cvtA(x, DIN);
    gemm(Wl0, bl0, nullptr, XL, DIN, HC);
    gemm(Wr0, br0, nullptr, XR, DIN, HC);
    gemm(Ws0, bs0, bc0,     X2, DIN, HC);             // agg init = skip + bs0 + bc0
    attention(XL, XR, att0, X2, Cc0, HC, 0, 1.0f);
    col_stats<<<HC, 256, 0, stream>>>(X2, HC, MU, VAR);
    bn_relu<<<(int)((elemsHC + 255) / 256), 256, 0, stream>>>(X2, g0, be0, MU, VAR, X1, HC);

    // ================= Layer 1:  x1[256] -> h[256] (pre-BN, returned) ========
    cvtA(X1, HC);
    gemm(Wl1, bl1, nullptr, XL, HC, HC);
    gemm(Wr1, br1, nullptr, XR, HC, HC);
    gemm(Ws1, bs1, bc1,     Hout, HC, HC);            // agg init into d_out's h
    attention(XL, XR, att1, Hout, Cc0, HC, 0, 1.0f);
    col_stats<<<HC, 256, 0, stream>>>(Hout, HC, MU, VAR);
    bn_relu<<<(int)((elemsHC + 255) / 256), 256, 0, stream>>>(Hout, g1, be1, MU, VAR, X2, HC);

    // ================= Layer 2:  x2[256] -> out[40] (head mean) =============
    cvtA(X2, HC);
    gemm(Wl2, bl2, nullptr, XL, HC, Hh * OUTW);       // N x 160
    gemm(Wr2, br2, nullptr, XR, HC, Hh * OUTW);
    gemm(Ws2, bs2, bc2,     Oout, HC, OUTW);          // agg init = skip + bs2 + bc2
    attention(XL, XR, att2, Oout, OUTW, OUTW, 1, 1.0f / (float)Hh);
}
